// GCNLayer_2869038154148
// MI455X (gfx1250) — matrix operand order repaired and checked by reading, not compiled, not run on hardware
//
#include <hip/hip_runtime.h>

// GCN layer for MI455X (gfx1250, wave32).
//   y = leaky_relu( Dhat^-1/2 (A+I) Dhat^-1/2 (X W) + b )
// Strategy: double-bf16 (hi+lo split) WMMA for both GEMMs, f32 accumulate:
//   C += Ah*Bh + Ah*Bl + Al*Bh  -> ~1e-5 relative error vs f32 reference.
// adj (134 MB) fits in the 192 MB L2; the degree pass warms L2 for the spmm.
// The spmm's B operand (XWs) is staged through LDS with async-to-LDS copies
// shared by all 8 waves of the workgroup (8x cut in L2-side B traffic).

typedef __attribute__((ext_vector_type(16))) __bf16 v16bf;
typedef __attribute__((ext_vector_type(8)))  float  v8f;
typedef __attribute__((ext_vector_type(4)))  int    v4i;

#define BB 8
#define NN 2048
#define FF 128

__device__ __forceinline__ void splitbf(float v, __bf16& h, __bf16& l) {
    h = (__bf16)v;
    l = (__bf16)(v - (float)h);
}

// ---------------------------------------------------------------------------
// Kernel 1: r[b,n] = 1/sqrt(1 + sum_m adj[b,n,m])   (the +1 is the self loop)
// One wave per row; coalesced float4 loads; wave32 xor-shuffle reduction.
// ---------------------------------------------------------------------------
__global__ __launch_bounds__(256) void k_rownorm(const float* __restrict__ adj,
                                                 float* __restrict__ r) {
    int wave = blockIdx.x * 8 + (threadIdx.x >> 5);
    int lane = threadIdx.x & 31;
    long row = wave;                         // 0 .. B*N-1
    const float* p = adj + row * (long)NN;
    float s = 0.0f;
    #pragma unroll 4
    for (int i = lane * 4; i < NN; i += 32 * 4) {
        float4 v = *(const float4*)(p + i);
        s += v.x + v.y + v.z + v.w;
    }
    #pragma unroll
    for (int off = 16; off > 0; off >>= 1)
        s += __shfl_xor(s, off, 32);
    if (lane == 0) r[row] = 1.0f / sqrtf(1.0f + s);
}

// ---------------------------------------------------------------------------
// Kernel 2: XWs[m,o] = r[m] * (X @ W)[m,o], stored TRANSPOSED [b][o][m]
// as split bf16 (hi array + lo array). One 16x16 tile per wave, K=128.
// ---------------------------------------------------------------------------
__global__ __launch_bounds__(256) void k_xw(const float* __restrict__ X,
                                            const float* __restrict__ W,
                                            const float* __restrict__ r,
                                            __bf16* __restrict__ xwh,
                                            __bf16* __restrict__ xwl) {
    int wave = blockIdx.x * 8 + (threadIdx.x >> 5);
    int lane = threadIdx.x & 31;
    int bb = wave >> 10;                 // / (128 m-tiles * 8 o-tiles)
    int mi = (wave >> 3) & 127;
    int oi = wave & 7;
    int m0 = mi << 4, o0 = oi << 4;
    int l15 = lane & 15, hh = lane >> 4;
    int abase = hh * 8;                  // A layout: lanes16-31 hold K=8..15,24..31
    int koff  = hh * 16;                 // unused here; B is read with abase, so lanes16-31 hold K=8..15,24..31 like A

    const float* arow = X + ((long)bb * NN + m0 + l15) * FF;
    v8f c = {};
    #pragma unroll
    for (int k0 = 0; k0 < FF; k0 += 32) {
        float va[16];
        *(float4*)(va + 0)  = *(const float4*)(arow + k0 + abase + 0);
        *(float4*)(va + 4)  = *(const float4*)(arow + k0 + abase + 4);
        *(float4*)(va + 8)  = *(const float4*)(arow + k0 + abase + 16);
        *(float4*)(va + 12) = *(const float4*)(arow + k0 + abase + 20);
        v16bf ah, al;
        #pragma unroll
        for (int i = 0; i < 16; ++i) { __bf16 h, l; splitbf(va[i], h, l); ah[i] = h; al[i] = l; }

        v16bf bh, bl;
        #pragma unroll
        for (int j = 0; j < 16; ++j) {
            float w = W[(long)(k0 + abase + (j & 7) + ((j >> 3) << 4)) * FF + (o0 + l15)];
            __bf16 h, l; splitbf(w, h, l); bh[j] = h; bl[j] = l;
        }
        c = __builtin_amdgcn_wmma_f32_16x16x32_bf16(false, ah, false, bh, (short)0, c, false, false);
        c = __builtin_amdgcn_wmma_f32_16x16x32_bf16(false, ah, false, bl, (short)0, c, false, false);
        c = __builtin_amdgcn_wmma_f32_16x16x32_bf16(false, al, false, bh, (short)0, c, false, false);
    }
    // scale by r[m], split to hi/lo bf16, store transposed [b][o][m]
    int o = o0 + l15;
    long colbase = ((long)bb * FF + o) * (long)NN;
    #pragma unroll
    for (int v = 0; v < 8; ++v) {
        int m = m0 + v + 8 * hh;                       // C layout: M = vgpr + 8*(lane>=16)
        float val = c[v] * r[(long)bb * NN + m];
        __bf16 h, l; splitbf(val, h, l);
        xwh[colbase + m] = h;
        xwl[colbase + m] = l;
    }
}

// ---------------------------------------------------------------------------
// Kernel 3: y[n,o] = leaky( r[n]*( adj[n,:] . XWs[:,o] + XWs[n,o] ) + b[o] )
// Block = 8 waves, each wave a 16-row n-band (128 rows/block) x 64 o-cols.
// B tiles (shared by all 8 waves) are staged through LDS with double-buffered
// GLOBAL_LOAD_ASYNC_TO_LDS_B128 + s_wait_asynccnt + barrier per 32-K step.
// ---------------------------------------------------------------------------
__global__ __launch_bounds__(256) void k_spmm(const float* __restrict__ adj,
                                              const __bf16* __restrict__ xwh,
                                              const __bf16* __restrict__ xwl,
                                              const float* __restrict__ r,
                                              const float* __restrict__ bias,
                                              float* __restrict__ out) {
    __shared__ __bf16 lds[2][2][64][32];   // [buffer][hi/lo][o-col][k] = 16 KB

    int tid  = threadIdx.x;
    int warp = tid >> 5, lane = tid & 31;
    int bid  = blockIdx.x;
    int bb   = bid >> 5;                  // 32 blocks per batch (16 n-blocks x 2 halves)
    int nblk = (bid >> 1) & 15;
    int half = bid & 1;
    int n0 = nblk * 128 + warp * 16;
    int o0 = half * 64;
    int l15 = lane & 15, hh = lane >> 4;
    int abase = hh * 16;
    int koff  = hh * 16;

    // cooperative async-copy mapping: thread -> (column, 8-halfword chunk)
    int ccol = tid >> 2;                  // 0..63
    int cchk = (tid & 3) * 8;             // halfword offset 0,8,16,24
    long gcol = ((long)bb * FF + (o0 + ccol)) * (long)NN + cchk;

    const float* arow = adj + ((long)bb * NN + n0 + l15) * (long)NN;
    v8f c[4] = {};

    // prologue: stage k0 = 0 into buffer 0
    __builtin_amdgcn_global_load_async_to_lds_b128(
        (v4i*)(xwh + gcol), (v4i*)&lds[0][0][ccol][cchk], 0, 0);
    __builtin_amdgcn_global_load_async_to_lds_b128(
        (v4i*)(xwl + gcol), (v4i*)&lds[0][1][ccol][cchk], 0, 0);

    for (int k0 = 0; k0 < NN; k0 += 32) {
        int cur = (k0 >> 5) & 1;
        __builtin_amdgcn_s_wait_asynccnt(0);   // this wave's copies into buf[cur] done
        __syncthreads();                       // everyone's copies visible; prev reads done
        if (k0 + 32 < NN) {
            int nxt = cur ^ 1;
            long g = gcol + k0 + 32;
            __builtin_amdgcn_global_load_async_to_lds_b128(
                (v4i*)(xwh + g), (v4i*)&lds[nxt][0][ccol][cchk], 0, 0);
            __builtin_amdgcn_global_load_async_to_lds_b128(
                (v4i*)(xwl + g), (v4i*)&lds[nxt][1][ccol][cchk], 0, 0);
        }

        __builtin_prefetch((const void*)(arow + k0 + 512), 0, 0);  // stream adj ahead
        float va[16];
        *(float4*)(va + 0)  = *(const float4*)(arow + k0 + abase + 0);
        *(float4*)(va + 4)  = *(const float4*)(arow + k0 + abase + 4);
        *(float4*)(va + 8)  = *(const float4*)(arow + k0 + abase + 8);
        *(float4*)(va + 12) = *(const float4*)(arow + k0 + abase + 12);
        v16bf ah, al;
        #pragma unroll
        for (int i = 0; i < 16; ++i) { __bf16 h, l; splitbf(va[i], h, l); ah[i] = h; al[i] = l; }

        #pragma unroll
        for (int t = 0; t < 4; ++t) {
            v16bf bhv = *(const v16bf*)&lds[cur][0][t * 16 + l15][koff];
            v16bf blv = *(const v16bf*)&lds[cur][1][t * 16 + l15][koff];
            c[t] = __builtin_amdgcn_wmma_f32_16x16x32_bf16(false, ah, false, bhv, (short)0, c[t], false, false);
            c[t] = __builtin_amdgcn_wmma_f32_16x16x32_bf16(false, ah, false, blv, (short)0, c[t], false, false);
            c[t] = __builtin_amdgcn_wmma_f32_16x16x32_bf16(false, al, false, bhv, (short)0, c[t], false, false);
        }
    }

    float rn[8];
    #pragma unroll
    for (int v = 0; v < 8; ++v) rn[v] = r[(long)bb * NN + n0 + v + 8 * hh];

    #pragma unroll
    for (int t = 0; t < 4; ++t) {
        int o = o0 + t * 16 + l15;
        float bo = bias[o];
        long cb = ((long)bb * FF + o) * (long)NN;
        #pragma unroll
        for (int v = 0; v < 8; ++v) {
            int n = n0 + v + 8 * hh;
            float self = (float)xwh[cb + n] + (float)xwl[cb + n];   // XWs[n,o] (identity term)
            float y = rn[v] * (c[t][v] + self) + bo;
            out[((long)bb * NN + n) * (long)FF + o] = fmaxf(y, 0.01f * y);  // leaky relu
        }
    }
}

// ---------------------------------------------------------------------------
extern "C" void kernel_launch(void* const* d_in, const int* in_sizes, int n_in,
                              void* d_out, int out_size, void* d_ws, size_t ws_size,
                              hipStream_t stream) {
    const float* X   = (const float*)d_in[0];   // [8,2048,128]
    const float* adj = (const float*)d_in[1];   // [8,2048,2048]
    const float* W   = (const float*)d_in[2];   // [128,128]
    const float* b   = (const float*)d_in[3];   // [128]
    float* out = (float*)d_out;                 // [8,2048,128] f32

    // workspace: r (64KB) | XWs_hi^T (4MB bf16) | XWs_lo^T (4MB bf16)
    char* ws = (char*)d_ws;
    float*  r   = (float*)ws;
    __bf16* xwh = (__bf16*)(ws + 65536);
    __bf16* xwl = (__bf16*)(ws + 65536 + (size_t)BB * FF * NN * 2);

    (void)in_sizes; (void)n_in; (void)out_size; (void)ws_size;

    k_rownorm<<<(BB * NN) / 8, 256, 0, stream>>>(adj, r);
    k_xw    <<<(BB * (NN / 16) * (FF / 16)) / 8, 256, 0, stream>>>(X, W, r, xwh, xwl);
    k_spmm  <<<BB * (NN / 128) * 2, 256, 0, stream>>>(adj, xwh, xwl, r, b, out);
}
